// MelPerformer_50491635532184
// MI455X (gfx1250) — matrix operand order repaired and checked by reading, not compiled, not run on hardware
//
#include <hip/hip_runtime.h>
#include <hip/hip_bf16.h>
#include <cstdint>
#include <cstddef>

// ---------------------------------------------------------------------------
// MI455X (gfx1250) MelPerformer forward, round 3.
// - All contractions K>=64 via v_wmma_f32_16x16x32_f16 (f16 in, f32 accum).
// - Every GEMM operand arranged K-contiguous: A[M][K] (row stride lda),
//   B as BT[N][K] (row stride ldb).  Tiles are both [64 rows x 32 halves].
// - Tile staging via the Tensor Data Mover (tensor_load_to_lds + TENSORcnt),
//   with TDM LDS padding producing 40-half padded rows (bank-conflict-free).
//   (6-arg clang-23 builtin form.)
// - Fragments = two ds_load_b128 per operand (K runs contiguous in LDS).
// ---------------------------------------------------------------------------

typedef _Float16 f16;
typedef _Float16 v16h __attribute__((ext_vector_type(16)));
typedef _Float16 v8h  __attribute__((ext_vector_type(8)));
typedef float    v8f  __attribute__((ext_vector_type(8)));
typedef unsigned int u32x4 __attribute__((ext_vector_type(4)));
typedef int          i32x8 __attribute__((ext_vector_type(8)));
typedef int          i32x4 __attribute__((ext_vector_type(4)));

// Issue one TDM 2-D tile load: 32 halves (64B) x 64 rows, row stride
// rowStrideElems (f16 elements), into LDS at lds_byte with +16B row padding
// (pad_interval=16 DWORDs -> code 3, pad_amount=4 DWORDs -> code 3),
// giving a 40-half padded LDS row pitch.
static __device__ __forceinline__ void tdm_load_tile(
    unsigned lds_byte, const f16* gaddr, long rowStrideElems)
{
  unsigned long long ga = (unsigned long long)(uintptr_t)gaddr;
  u32x4 g0;
  g0[0] = 1u;                                      // count=1, user descriptor
  g0[1] = lds_byte;                                // lds_addr (bytes)
  g0[2] = (unsigned)(ga & 0xffffffffu);            // global_addr[31:0]
  g0[3] = (unsigned)((ga >> 32) & 0x1ffffffu)      // global_addr[56:32]
        | (2u << 30);                              // type=2 ("image")
  i32x8 g1;
  g1[0] = (1 << 16)                                // data_size = 2 bytes
        | (1 << 20)                                // pad_enable
        | (3 << 22)                                // pad_interval: 16 DWORDs
        | (3 << 25);                               // pad_amount:   4 DWORDs
  g1[1] = 0;                                       // no atomic barrier; dim0 lo16=0
  g1[2] = 0x4000;                                  // tensor_dim0 = 2^30 (no clip)
  g1[3] = 0x4000 | (32 << 16);                     // tensor_dim1 = 2^30; tile_dim0=32
  g1[4] = 64;                                      // tile_dim1=64, tile_dim2=0 (2D)
  unsigned long long st = (unsigned long long)rowStrideElems;
  g1[5] = (int)(st & 0xffffffffu);                 // tensor_dim0_stride[31:0]
  g1[6] = (int)((st >> 32) & 0xffffu);             // stride[47:32]; dim1_stride=0
  g1[7] = 0;
  i32x4 z4 = {0, 0, 0, 0};                         // groups 2/3 unused (2-D tensor)
  i32x8 z8 = {0, 0, 0, 0, 0, 0, 0, 0};
  __builtin_amdgcn_tensor_load_to_lds(g0, g1, z4, z4, z8, 0);
}

// ======================= WMMA tiled GEMM (TDM-fed) =========================
// C[M,N] = epilogue( A[M,K] * BT[N,K]^T ); out index = cOff + row*ldcR + col*ldcC.
// Batched via gridDim.z (outer=b, inner=h).  128 threads = 4 waves;
// block tile 64x64, K-step 32; wave w owns rows [16w,16w+16) x 64 cols.
__global__ __launch_bounds__(128) void k_wmma_gemm(
    const f16* __restrict__ A, const f16* __restrict__ Bt,
    float* __restrict__ o32, f16* __restrict__ o16,
    const float* __restrict__ colScale, const float* __restrict__ colBias,
    const float* __restrict__ rowScale, const float* __restrict__ resid,
    int M, int N, int K,
    long lda, long ldb, long ldcR, long ldcC,
    float sscale, int act, int batchInner,
    long aBo, long aBi, long bBo, long bBi, long cBo, long cBi, long rsB)
{
  __shared__ f16 smem[2 * 64 * 40];                // A tile | B tile, padded rows
  f16* As = smem;
  f16* Bs = smem + 64 * 40;

  const int bz    = blockIdx.z;
  const int outer = bz / batchInner;
  const int inner = bz - outer * batchInner;
  const f16* Ab = A  + (long)outer*aBo + (long)inner*aBi;
  const f16* Bb = Bt + (long)outer*bBo + (long)inner*bBi;
  const long cOff  = (long)outer*cBo + (long)inner*cBi;
  const long rsOff = (long)bz * rsB;

  const int row0 = blockIdx.y * 64, col0 = blockIdx.x * 64;
  const int tid = threadIdx.x, lane = tid & 31, wave = tid >> 5;

  v8f acc[4];
  #pragma unroll
  for (int i = 0; i < 4; i++)
    #pragma unroll
    for (int j = 0; j < 8; j++) acc[i][j] = 0.f;

  const unsigned ldsA = (unsigned)(uintptr_t)As;   // flat LDS addr low 32 = offset
  const unsigned ldsB = (unsigned)(uintptr_t)Bs;

  for (int k0 = 0; k0 < K; k0 += 32) {
    if (tid < 32) {                                // wave 0 drives the TDM
      tdm_load_tile(ldsA, Ab + (long)row0*lda + k0, lda);
      tdm_load_tile(ldsB, Bb + (long)col0*ldb + k0, ldb);
      __builtin_amdgcn_s_wait_tensorcnt(0);
    }
    __syncthreads();

    // A fragment: lanes 0-15 row M=lane K{kh..kh+7, kh+16..kh+23}, kh=0;
    // lanes 16-31 same rows, kh=8 (ISA 7.12.2).  Two 16B LDS loads.
    const int r  = lane & 15;
    const int kh = (lane >> 4) * 8;
    const f16* arow = As + (wave*16 + r) * 40;
    v8h alo = *(const v8h*)(arow + kh);
    v8h ahi = *(const v8h*)(arow + kh + 16);
    v16h af = __builtin_shufflevector(alo, ahi,
        0,1,2,3,4,5,6,7,8,9,10,11,12,13,14,15);

    #pragma unroll
    for (int ct = 0; ct < 4; ct++) {
      // B fragment: lane<16 -> col N=lane, K{kh..kh+7, kh+16..kh+23} (kh = (lane>=16)*8);
      // BT tile is [n][k] so each 8 K-half run is contiguous: two 16B loads.
      const f16* brow = Bs + (ct*16 + (lane & 15)) * 40 + (lane >> 4) * 8;
      v8h blo = *(const v8h*)brow;
      v8h bhi = *(const v8h*)(brow + 16);
      v16h bf = __builtin_shufflevector(blo, bhi,
          0,1,2,3,4,5,6,7,8,9,10,11,12,13,14,15);
      acc[ct] = __builtin_amdgcn_wmma_f32_16x16x32_f16(
          false, af, false, bf, (short)0, acc[ct], false, false);
    }
    __syncthreads();
  }

  // Epilogue. C layout: vgpr i -> M = 16w + (lane>=16?8:0) + i, N = lane&15.
  const int r8 = (lane >> 4) * 8, cl = lane & 15;
  #pragma unroll
  for (int ct = 0; ct < 4; ct++) {
    const int col = col0 + ct*16 + cl;
    #pragma unroll
    for (int i = 0; i < 8; i++) {
      const int row = row0 + wave*16 + r8 + i;
      float v = acc[ct][i];
      if (colScale) v *= colScale[col];
      if (colBias)  v += colBias[col];
      v *= sscale;
      if (rowScale) v *= rowScale[rsOff + row];
      if (act == 1)      v = fmaxf(v, 0.f);
      else if (act == 2) v = fmaxf(v, 0.f) + 1e-3f;
      const long oi = cOff + (long)row*ldcR + (long)col*ldcC;
      if (resid) v += resid[oi];
      if (o32) o32[oi] = v;
      if (o16) o16[oi] = (f16)v;
    }
  }
}

// ============================ helper kernels ===============================
__global__ __launch_bounds__(256) void k_convert(const float* __restrict__ s,
                                                 f16* __restrict__ d, long n)
{
  long i = blockIdx.x * (long)blockDim.x + threadIdx.x;
  if (i < n) d[i] = (f16)s[i];
}

// transpose-convert: src f32 [L][K][N] -> dst f16 [L][N][K]
__global__ __launch_bounds__(256) void k_convt(const float* __restrict__ s,
                                               f16* __restrict__ d,
                                               int K, int N, int Lb)
{
  long i = blockIdx.x * (long)blockDim.x + threadIdx.x;
  long tot = (long)Lb * K * N;
  if (i >= tot) return;
  int k = (int)(i % K);
  int n = (int)((i / K) % N);
  int l = (int)(i / ((long)K * N));
  d[i] = (f16)s[((long)l*K + k)*N + n];
}

// fused: conv0 (1x1, per-channel scale) + depthwise 3x3 + bias -> d1[b,c,f,t]
__global__ __launch_bounds__(256) void k_front_dw1(
    const float* __restrict__ x, const float* __restrict__ w0,
    const float* __restrict__ w1, const float* __restrict__ b1,
    float* __restrict__ d1, int T, int NM, int Bn)
{
  long i = blockIdx.x * (long)blockDim.x + threadIdx.x;
  long n = (long)Bn * 4 * NM * T;
  if (i >= n) return;
  int t = (int)(i % T);
  int f = (int)((i / T) % NM);
  int c = (int)((i / ((long)T * NM)) % 4);
  int b = (int)(i / ((long)T * NM * 4));
  float acc = b1[c];
  float s0 = w0[c];
  for (int df = 0; df < 3; df++) {
    int f2 = f + df - 1;
    if (f2 < 0 || f2 >= NM) continue;
    for (int dt = 0; dt < 3; dt++) {
      int t2 = t + dt - 1;
      if (t2 < 0 || t2 >= T) continue;
      acc += x[((long)b*T + t2)*NM + f2] * s0 * w1[(c*3 + df)*3 + dt];
    }
  }
  d1[i] = acc;
}

// pw1 (K=4) + bn1 + relu -> p1[b,f,t,o] (f16)
__global__ __launch_bounds__(256) void k_pw1(
    const float* __restrict__ d1, const float* __restrict__ w,
    const float* __restrict__ bias,
    const float* __restrict__ g, const float* __restrict__ bb,
    const float* __restrict__ m, const float* __restrict__ v,
    f16* __restrict__ p1, int T, int NM, int Bn)
{
  long i = blockIdx.x * (long)blockDim.x + threadIdx.x;
  long n = (long)Bn * NM * T * 128;
  if (i >= n) return;
  int o = (int)(i % 128);
  int t = (int)((i / 128) % T);
  int f = (int)((i / (128L * T)) % NM);
  int b = (int)(i / (128L * T * NM));
  float acc = bias[o];
  for (int c = 0; c < 4; c++)
    acc += d1[(((long)b*4 + c)*NM + f)*T + t] * w[o*4 + c];
  float rs = rsqrtf(v[o] + 1e-5f);
  float vv = (acc - m[o]) * rs * g[o] + bb[o];
  p1[i] = (f16)fmaxf(vv, 0.f);
}

// depthwise 3x3 on 128 ch + bias -> d2[b,f,t,c] (f16)
__global__ __launch_bounds__(256) void k_dw2(
    const f16* __restrict__ p1, const float* __restrict__ w,
    const float* __restrict__ bias, f16* __restrict__ d2,
    int T, int NM, int Bn)
{
  long i = blockIdx.x * (long)blockDim.x + threadIdx.x;
  long n = (long)Bn * NM * T * 128;
  if (i >= n) return;
  int c = (int)(i % 128);
  int t = (int)((i / 128) % T);
  int f = (int)((i / (128L * T)) % NM);
  int b = (int)(i / (128L * T * NM));
  float acc = bias[c];
  for (int df = 0; df < 3; df++) {
    int f2 = f + df - 1;
    if (f2 < 0 || f2 >= NM) continue;
    for (int dt = 0; dt < 3; dt++) {
      int t2 = t + dt - 1;
      if (t2 < 0 || t2 >= T) continue;
      acc += (float)p1[(((long)b*NM + f2)*T + t2)*128 + c] * w[c*9 + df*3 + dt];
    }
  }
  d2[i] = (f16)acc;
}

// fold bn2 + conv bias into per-column scale/bias for the pw2 GEMM epilogue
__global__ __launch_bounds__(256) void k_bnfold(
    const float* g, const float* b, const float* m, const float* v,
    const float* convb, float* cs, float* cb, int C)
{
  int o = threadIdx.x;
  if (o >= C) return;
  float s = g[o] * rsqrtf(v[o] + 1e-5f);
  cs[o] = s;
  cb[o] = (convb[o] - m[o]) * s + b[o];
}

// mean over mel axis: pw2o[b,f,t,c] -> hmean[b,t,c] f16
__global__ __launch_bounds__(256) void k_mean(
    const f16* __restrict__ src, f16* __restrict__ dst, int T, int NM, int Bn)
{
  long i = blockIdx.x * (long)blockDim.x + threadIdx.x;
  long n = (long)Bn * T * 256;
  if (i >= n) return;
  int c = (int)(i % 256);
  int t = (int)((i / 256) % T);
  int b = (int)(i / (256L * T));
  float s = 0.f;
  for (int f = 0; f < NM; f++)
    s += (float)src[(((long)b*NM + f)*T + t)*256 + c];
  dst[i] = (f16)(s / (float)NM);
}

// LN + pos_emb, fan out to both reversible streams
__global__ __launch_bounds__(256) void k_ln0_pos(
    const float* __restrict__ in, const float* __restrict__ g,
    const float* __restrict__ b, const float* __restrict__ pos,
    float* __restrict__ x1, float* __restrict__ x2, int R, int C, int Tmod)
{
  int row = (int)((blockIdx.x * (long)blockDim.x + threadIdx.x) >> 5);
  if (row >= R) return;
  int lane = threadIdx.x & 31;
  const float* r = in + (long)row * C;
  float s = 0.f, s2 = 0.f;
  for (int c = lane; c < C; c += 32) { float v = r[c]; s += v; s2 += v*v; }
  for (int m = 16; m; m >>= 1) { s += __shfl_xor(s, m); s2 += __shfl_xor(s2, m); }
  float mu = s / C, var = s2 / C - mu*mu, rs = rsqrtf(var + 1e-5f);
  const float* p = pos + (long)(row % Tmod) * C;
  for (int c = lane; c < C; c += 32) {
    float v = (r[c] - mu) * rs * g[c] + b[c] + p[c];
    long o = (long)row * C + c;
    x1[o] = v; x2[o] = v;
  }
}

// generic layernorm (wave per row), optional relu, f32 and/or f16 outputs
__global__ __launch_bounds__(256) void k_layernorm(
    const float* __restrict__ in, const float* __restrict__ g,
    const float* __restrict__ b, float* __restrict__ o32,
    f16* __restrict__ o16, int R, int C, int act)
{
  int row = (int)((blockIdx.x * (long)blockDim.x + threadIdx.x) >> 5);
  if (row >= R) return;
  int lane = threadIdx.x & 31;
  const float* r = in + (long)row * C;
  float s = 0.f, s2 = 0.f;
  for (int c = lane; c < C; c += 32) { float v = r[c]; s += v; s2 += v*v; }
  for (int m = 16; m; m >>= 1) { s += __shfl_xor(s, m); s2 += __shfl_xor(s2, m); }
  float mu = s / C, var = s2 / C - mu*mu, rs = rsqrtf(var + 1e-5f);
  for (int c = lane; c < C; c += 32) {
    float v = (r[c] - mu) * rs * g[c] + b[c];
    if (act) v = fmaxf(v, 0.f);
    long o = (long)row * C + c;
    if (o32) o32[o] = v;
    if (o16) o16[o] = (f16)v;
  }
}

// kf_sum[b,h,j] = sum_n kfT[b,h,j,n]   (contiguous inner loop)
__global__ __launch_bounds__(256) void k_kfsum(
    const f16* __restrict__ kfT, float* __restrict__ kfs,
    int BH, int Mf, int N)
{
  long i = blockIdx.x * (long)blockDim.x + threadIdx.x;
  long tot = (long)BH * Mf;
  if (i >= tot) return;
  const f16* p = kfT + i * (long)N;
  float s = 0.f;
  for (int n = 0; n < N; n++) s += (float)p[n];
  kfs[i] = s;
}

// d_inv[b,h,n] = 1 / dot(qf[b,n,h,:], kf_sum[b,h,:])
__global__ __launch_bounds__(256) void k_dinv(
    const f16* __restrict__ qf, const float* __restrict__ kfs,
    float* __restrict__ dinv, long BHN, int Mf, int N, int H)
{
  long i = blockIdx.x * (long)blockDim.x + threadIdx.x;
  if (i >= BHN) return;
  int n = (int)(i % N);
  int bh = (int)(i / N);
  int h = bh % H, b = bh / H;
  const f16* q = qf + ((((long)b * N) + n) * H + h) * Mf;
  const float* ks = kfs + (long)bh * Mf;
  float s = 0.f;
  for (int j = 0; j < Mf; j++) s += (float)q[j] * ks[j];
  dinv[i] = 1.f / s;
}

// GLU: g = gelu_exact(u[:, :Hh]) * u[:, Hh:2Hh]  -> f16
__global__ __launch_bounds__(256) void k_glu(
    const float* __restrict__ u, f16* __restrict__ g, long n, int Hh, int W)
{
  long i = blockIdx.x * (long)blockDim.x + threadIdx.x;
  if (i >= n) return;
  long row = i / Hh;
  int c = (int)(i % Hh);
  float u1 = u[row * W + c];
  float u2 = u[row * W + Hh + c];
  float ge = 0.5f * u1 * (1.f + erff(u1 * 0.70710678118654752f));
  g[i] = (f16)(ge * u2);
}

__global__ __launch_bounds__(256) void k_addpair(
    const float* __restrict__ a, const float* __restrict__ b,
    f16* __restrict__ o, long n)
{
  long i = blockIdx.x * (long)blockDim.x + threadIdx.x;
  if (i < n) o[i] = (f16)(a[i] + b[i]);
}

// out[row] = dot(cact[row,:], w2) + b2   (wave per row)
__global__ __launch_bounds__(256) void k_clf_out(
    const float* __restrict__ cact, const float* __restrict__ w2,
    const float* __restrict__ b2, float* __restrict__ out, int R, int C)
{
  int row = (int)((blockIdx.x * (long)blockDim.x + threadIdx.x) >> 5);
  if (row >= R) return;
  int lane = threadIdx.x & 31;
  float s = 0.f;
  for (int c = lane; c < C; c += 32) s += cact[(long)row * C + c] * w2[c];
  for (int m = 16; m; m >>= 1) s += __shfl_xor(s, m);
  if (lane == 0) out[row] = s + b2[0];
}

// ============================== host side ==================================
static inline unsigned cdiv(long a, long b) { return (unsigned)((a + b - 1) / b); }

static void launch_gemm(hipStream_t st,
    const f16* A, const f16* Bt, float* o32, f16* o16,
    const float* cs, const float* cb, const float* rs, const float* res,
    int M, int N, int K,
    long lda, long ldb, long ldcR, long ldcC,
    float ss, int act, int nBatch, int bInner,
    long aBo, long aBi, long bBo, long bBi, long cBo, long cBi, long rsB)
{
  dim3 g(N / 64, M / 64, nBatch), b(128, 1, 1);
  k_wmma_gemm<<<g, b, 0, st>>>(A, Bt, o32, o16, cs, cb, rs, res, M, N, K,
                               lda, ldb, ldcR, ldcC, ss, act, bInner,
                               aBo, aBi, bBo, bBi, cBo, cBi, rsB);
}

extern "C" void kernel_launch(void* const* d_in, const int* in_sizes, int n_in,
                              void* d_out, int out_size, void* d_ws, size_t ws_size,
                              hipStream_t stream)
{
  constexpr int Bn = 2, T = 2048, NM = 64, D = 256, H = 4, DH = 64;
  constexpr int MF = 384, L = 4, FF2 = 2048, FFH = 1024;
  constexpr int RT = Bn * T;
  const float dn = 0.35355339059327373f;    // DH^-0.25

  const float* x       = (const float*)d_in[0];
  const float* conv0_w = (const float*)d_in[1];
  const float* dw1_w   = (const float*)d_in[2];
  const float* dw1_b   = (const float*)d_in[3];
  const float* pw1_w   = (const float*)d_in[4];
  const float* pw1_b   = (const float*)d_in[5];
  const float* bn1_g   = (const float*)d_in[6];
  const float* bn1_b   = (const float*)d_in[7];
  const float* bn1_m   = (const float*)d_in[8];
  const float* bn1_v   = (const float*)d_in[9];
  const float* dw2_w   = (const float*)d_in[10];
  const float* dw2_b   = (const float*)d_in[11];
  const float* pw2_w   = (const float*)d_in[12];
  const float* pw2_b   = (const float*)d_in[13];
  const float* bn2_g   = (const float*)d_in[14];
  const float* bn2_b   = (const float*)d_in[15];
  const float* bn2_m   = (const float*)d_in[16];
  const float* bn2_v   = (const float*)d_in[17];
  const float* proj_w  = (const float*)d_in[18];
  const float* proj_b  = (const float*)d_in[19];
  const float* ln0_g   = (const float*)d_in[20];
  const float* ln0_b   = (const float*)d_in[21];
  const float* pos_emb = (const float*)d_in[22];
  const float* ln1_g   = (const float*)d_in[23];
  const float* ln1_b   = (const float*)d_in[24];
  const float* wq      = (const float*)d_in[25];
  const float* bq      = (const float*)d_in[26];
  const float* wk      = (const float*)d_in[27];
  const float* bk      = (const float*)d_in[28];
  const float* wv      = (const float*)d_in[29];
  const float* bv      = (const float*)d_in[30];
  const float* wo      = (const float*)d_in[31];
  const float* bo      = (const float*)d_in[32];
  const float* proj_mat= (const float*)d_in[33];
  const float* ln2_g   = (const float*)d_in[34];
  const float* ln2_b   = (const float*)d_in[35];
  const float* ffw1    = (const float*)d_in[36];
  const float* ffb1    = (const float*)d_in[37];
  const float* ffw2    = (const float*)d_in[38];
  const float* ffb2    = (const float*)d_in[39];
  const float* clf_w1  = (const float*)d_in[40];
  const float* clf_b1  = (const float*)d_in[41];
  const float* clf_ln_g= (const float*)d_in[42];
  const float* clf_ln_b= (const float*)d_in[43];
  const float* clf_w2  = (const float*)d_in[44];
  const float* clf_b2  = (const float*)d_in[45];

  uint8_t* wp = (uint8_t*)d_ws;
  auto alloc = [&](size_t bytes) -> void* {
    void* p = (void*)wp;
    wp += (bytes + 255) & ~(size_t)255;
    return p;
  };

  // f16 weights: B operands stored as BT[N][K] (K contiguous)
  f16* pw2w16  = (f16*)alloc((size_t)256*128*2);     // [O=256][I=128] native
  f16* projT16 = (f16*)alloc((size_t)D*D*2);
  f16* wqT16   = (f16*)alloc((size_t)L*D*D*2);
  f16* wkT16   = (f16*)alloc((size_t)L*D*D*2);
  f16* wvT16   = (f16*)alloc((size_t)L*D*D*2);
  f16* woT16   = (f16*)alloc((size_t)L*D*D*2);
  f16* pm16    = (f16*)alloc((size_t)L*MF*DH*2);     // [j=384][d=64] native
  f16* ffw1T16 = (f16*)alloc((size_t)L*D*FF2*2);     // [2048][256]
  f16* ffw2T16 = (f16*)alloc((size_t)L*FFH*D*2);     // [256][1024]
  f16* clfw1T16= (f16*)alloc((size_t)D*128*2);       // [128][256]

  // activations
  float* d1     = (float*)alloc((size_t)Bn*4*NM*T*4);
  f16*   p1_16  = (f16*)  alloc((size_t)Bn*NM*T*128*2);
  f16*   d2_16  = (f16*)  alloc((size_t)Bn*NM*T*128*2);
  float* cs2    = (float*)alloc(256*4);
  float* cb2    = (float*)alloc(256*4);
  f16*   pw2o16 = (f16*)  alloc((size_t)Bn*NM*T*256*2);
  f16*   hmean16= (f16*)  alloc((size_t)RT*D*2);
  float* h0     = (float*)alloc((size_t)RT*D*4);
  float* x1     = (float*)alloc((size_t)RT*D*4);
  float* x2     = (float*)alloc((size_t)RT*D*4);
  f16*   a16    = (f16*)  alloc((size_t)RT*D*2);
  f16*   q16    = (f16*)  alloc((size_t)RT*D*2);
  f16*   k16    = (f16*)  alloc((size_t)RT*D*2);
  f16*   vT16   = (f16*)  alloc((size_t)Bn*H*DH*T*2);   // [b,h,d,n]
  f16*   qf16   = (f16*)  alloc((size_t)RT*H*MF*2);     // [b,n,h,j]
  f16*   kfT16  = (f16*)  alloc((size_t)RT*H*MF*2);     // [b,h,j,n]
  float* kfs    = (float*)alloc((size_t)Bn*H*MF*4);
  float* dinv   = (float*)alloc((size_t)Bn*H*T*4);
  f16*   ctxT16 = (f16*)  alloc((size_t)Bn*H*MF*DH*2);  // [b,h,d,j]
  f16*   o16buf = (f16*)  alloc((size_t)RT*D*2);        // [b,n,h,d]
  f16*   fb16   = (f16*)  alloc((size_t)RT*D*2);
  float* ubuf   = (float*)alloc((size_t)RT*FF2*4);
  f16*   g16    = (f16*)  alloc((size_t)RT*FFH*2);
  f16*   hsum16 = (f16*)  alloc((size_t)RT*D*2);
  float* ctmp   = (float*)alloc((size_t)RT*128*4);
  float* cact   = (float*)alloc((size_t)RT*128*4);
  (void)ws_size; (void)n_in; (void)in_sizes; (void)out_size;

  auto conv = [&](const float* s, f16* dst, long n) {
    k_convert<<<cdiv(n, 256), 256, 0, stream>>>(s, dst, n);
  };
  auto convt = [&](const float* s, f16* dst, int K, int N, int Lb) {
    k_convt<<<cdiv((long)Lb*K*N, 256), 256, 0, stream>>>(s, dst, K, N, Lb);
  };
  conv (pw2_w,   pw2w16,  (long)256*128);   // already [N][K]
  conv (proj_mat,pm16,    (long)L*MF*DH);   // already [N][K]
  convt(proj_w,  projT16, D, D, 1);
  convt(wq,      wqT16,   D, D, L);
  convt(wk,      wkT16,   D, D, L);
  convt(wv,      wvT16,   D, D, L);
  convt(wo,      woT16,   D, D, L);
  convt(ffw1,    ffw1T16, D, FF2, L);
  convt(ffw2,    ffw2T16, FFH, D, L);
  convt(clf_w1,  clfw1T16,D, 128, 1);

  // ---------------- conv frontend ----------------
  k_front_dw1<<<cdiv((long)Bn*4*NM*T, 256), 256, 0, stream>>>(
      x, conv0_w, dw1_w, dw1_b, d1, T, NM, Bn);
  k_pw1<<<cdiv((long)Bn*NM*T*128, 256), 256, 0, stream>>>(
      d1, pw1_w, pw1_b, bn1_g, bn1_b, bn1_m, bn1_v, p1_16, T, NM, Bn);
  k_dw2<<<cdiv((long)Bn*NM*T*128, 256), 256, 0, stream>>>(
      p1_16, dw2_w, dw2_b, d2_16, T, NM, Bn);
  k_bnfold<<<1, 256, 0, stream>>>(bn2_g, bn2_b, bn2_m, bn2_v, pw2_b, cs2, cb2, 256);

  // pw2 as GEMM: [B*NM*T,128] x [128->256], fused bn2+relu, f16 out
  launch_gemm(stream, d2_16, pw2w16, nullptr, pw2o16, cs2, cb2, nullptr, nullptr,
              Bn*NM*T, 256, 128, 128, 128, 256, 1,
              1.f, 1, 1, 1, 0,0,0,0,0,0,0);
  k_mean<<<cdiv((long)RT*256, 256), 256, 0, stream>>>(pw2o16, hmean16, T, NM, Bn);

  launch_gemm(stream, hmean16, projT16, h0, nullptr, nullptr, proj_b, nullptr, nullptr,
              RT, D, D, D, D, D, 1, 1.f, 0, 1, 1, 0,0,0,0,0,0,0);
  k_ln0_pos<<<cdiv((long)RT*32, 256), 256, 0, stream>>>(
      h0, ln0_g, ln0_b, pos_emb, x1, x2, RT, D, T);

  // ---------------- reversible Performer layers ----------------
  for (int l = 0; l < L; l++) {
    const f16* wqTl = wqT16 + (long)l*D*D;
    const f16* wkTl = wkT16 + (long)l*D*D;
    const f16* wvTl = wvT16 + (long)l*D*D;
    const f16* woTl = woT16 + (long)l*D*D;
    const f16* pml  = pm16  + (long)l*MF*DH;
    const f16* f1l  = ffw1T16 + (long)l*D*FF2;
    const f16* f2l  = ffw2T16 + (long)l*FFH*D;

    k_layernorm<<<cdiv((long)RT*32, 256), 256, 0, stream>>>(
        x2, ln1_g + l*D, ln1_b + l*D, nullptr, a16, RT, D, 0);

    // q,k (scaled by dn) -> f16 [b,n,h,d]
    launch_gemm(stream, a16, wqTl, nullptr, q16, nullptr, bq + l*D, nullptr, nullptr,
                RT, D, D, D, D, D, 1, dn, 0, 1, 1, 0,0,0,0,0,0,0);
    launch_gemm(stream, a16, wkTl, nullptr, k16, nullptr, bk + l*D, nullptr, nullptr,
                RT, D, D, D, D, D, 1, dn, 0, 1, 1, 0,0,0,0,0,0,0);
    // v, batched over b, written transposed: vT[b,h,d,n]
    launch_gemm(stream, a16, wvTl, nullptr, vT16, nullptr, bv + l*D, nullptr, nullptr,
                T, D, D, D, D, /*ldcR*/1, /*ldcC*/T, 1.f, 0, Bn, 1,
                /*aBo*/(long)T*D, 0, 0, 0, /*cBo*/(long)H*DH*T, 0, 0);

    // qf = relu(q @ P^T)+1e-3, batched (b,h): A rows n (lda=256), out [b,n,h,j]
    launch_gemm(stream, q16, pml, nullptr, qf16, nullptr, nullptr, nullptr, nullptr,
                T, MF, DH, /*lda*/(long)H*DH, /*ldb*/DH, /*ldcR*/(long)H*MF, 1,
                1.f, 2, Bn*H, H,
                /*aBo*/(long)T*D, /*aBi*/DH, 0, 0,
                /*cBo*/(long)T*H*MF, /*cBi*/MF, 0);
    // kf, written transposed: kfT[b,h,j,n]
    launch_gemm(stream, k16, pml, nullptr, kfT16, nullptr, nullptr, nullptr, nullptr,
                T, MF, DH, (long)H*DH, DH, /*ldcR*/1, /*ldcC*/T,
                1.f, 2, Bn*H, H,
                (long)T*D, DH, 0, 0,
                /*cBo*/(long)H*MF*T, /*cBi*/(long)MF*T, 0);

    k_kfsum<<<cdiv((long)Bn*H*MF, 256), 256, 0, stream>>>(kfT16, kfs, Bn*H, MF, T);
    k_dinv<<<cdiv((long)Bn*H*T, 256), 256, 0, stream>>>(
        qf16, kfs, dinv, (long)Bn*H*T, MF, T, H);

    // ctx = kf^T @ v, batched (b,h): A=kfT [j][n], B=vT [d][n]; out ctxT[b,h,d,j]
    launch_gemm(stream, kfT16, vT16, nullptr, ctxT16, nullptr, nullptr, nullptr, nullptr,
                MF, DH, T, /*lda*/T, /*ldb*/T, /*ldcR*/1, /*ldcC*/MF,
                1.f, 0, Bn*H, H,
                /*aBo*/(long)H*MF*T, /*aBi*/(long)MF*T,
                /*bBo*/(long)H*DH*T, /*bBi*/(long)DH*T,
                /*cBo*/(long)H*DH*MF, /*cBi*/(long)DH*MF, 0);

    // o = d_inv * (qf @ ctx): A=qf [n][j], B=ctxT [d][j]; out o[b,n,h,d]
    launch_gemm(stream, qf16, ctxT16, nullptr, o16buf, nullptr, nullptr, dinv, nullptr,
                T, DH, MF, /*lda*/(long)H*MF, /*ldb*/MF, /*ldcR*/D, 1,
                1.f, 0, Bn*H, H,
                /*aBo*/(long)T*H*MF, /*aBi*/MF,
                /*bBo*/(long)H*DH*MF, /*bBi*/(long)DH*MF,
                /*cBo*/(long)T*D, /*cBi*/DH, /*rsB*/T);

    // x1 += o @ wo + bo
    launch_gemm(stream, o16buf, woTl, x1, nullptr, nullptr, bo + l*D, nullptr, x1,
                RT, D, D, D, D, D, 1, 1.f, 0, 1, 1, 0,0,0,0,0,0,0);

    // f = LN(x1); u = f @ ffw1 + b; x2 += (gelu(u1)*u2) @ ffw2 + b
    k_layernorm<<<cdiv((long)RT*32, 256), 256, 0, stream>>>(
        x1, ln2_g + l*D, ln2_b + l*D, nullptr, fb16, RT, D, 0);
    launch_gemm(stream, fb16, f1l, ubuf, nullptr, nullptr, ffb1 + l*FF2, nullptr, nullptr,
                RT, FF2, D, D, D, FF2, 1, 1.f, 0, 1, 1, 0,0,0,0,0,0,0);
    k_glu<<<cdiv((long)RT*FFH, 256), 256, 0, stream>>>(
        ubuf, g16, (long)RT*FFH, FFH, FF2);
    launch_gemm(stream, g16, f2l, x2, nullptr, nullptr, ffb2 + l*D, nullptr, x2,
                RT, D, FFH, FFH, FFH, D, 1, 1.f, 0, 1, 1, 0,0,0,0,0,0,0);
  }

  // ---------------- classifier ----------------
  k_addpair<<<cdiv((long)RT*D, 256), 256, 0, stream>>>(x1, x2, hsum16, (long)RT*D);
  launch_gemm(stream, hsum16, clfw1T16, ctmp, nullptr, nullptr, clf_b1, nullptr, nullptr,
              RT, 128, D, D, D, 128, 1, 1.f, 0, 1, 1, 0,0,0,0,0,0,0);
  k_layernorm<<<cdiv((long)RT*32, 256), 256, 0, stream>>>(
      ctmp, clf_ln_g, clf_ln_b, cact, nullptr, RT, 128, 1);
  k_clf_out<<<cdiv((long)RT*32, 256), 256, 0, stream>>>(
      cact, clf_w2, clf_b2, (float*)d_out, RT, 128);
}
